// STAMPEDE_Wrapper_13941463842837
// MI455X (gfx1250) — matrix operand order repaired and checked by reading, not compiled, not run on hardware
//
#include <hip/hip_runtime.h>
#include <math.h>

// ---------------------------------------------------------------------------
// TGAT forward for MI455X (gfx1250): all GEMMs via v_wmma_f32_16x16x32_bf16.
// ---------------------------------------------------------------------------

typedef __bf16 bf16_t;
typedef bf16_t          v16bf __attribute__((ext_vector_type(16)));
typedef float           v8f   __attribute__((ext_vector_type(8)));
typedef unsigned short  v8us  __attribute__((ext_vector_type(8)));
typedef unsigned short  v16us __attribute__((ext_vector_type(16)));

__device__ __forceinline__ unsigned short f2bf(float f) {
    unsigned int u = __float_as_uint(f);
    u += 0x7FFFu + ((u >> 16) & 1u);          // round-to-nearest-even
    return (unsigned short)(u >> 16);
}
__device__ __forceinline__ float bf2f(unsigned short h) {
    return __uint_as_float(((unsigned int)h) << 16);
}

// Two 16B loads -> one 32B bf16 fragment
__device__ __forceinline__ v16bf load_frag16(const unsigned short* p0,
                                             const unsigned short* p1) {
    union { v8us h[2]; v16us w; } u;
    u.h[0] = *(const v8us*)p0;
    u.h[1] = *(const v8us*)p1;
    return __builtin_bit_cast(v16bf, u.w);
}

// A fragment: 16x32 bf16 tile (row-major in LDS, row stride lda elems).
// lane m=l&15, half=l>>4: VGPR g<4 -> k = half*8 + 2g{+1}; g>=4 -> k = 16 + half*8 + 2(g-4){+1}
__device__ __forceinline__ v16bf load_a_frag(const unsigned short* a, int lda) {
    int l = (int)(threadIdx.x & 31);
    int m = l & 15, half = l >> 4;
    const unsigned short* base = a + m * lda + half * 8;
    return load_frag16(base, base + 16);
}

// B fragment: weights stored transposed/N-major: Wt[Npad][Kpad].
// lane n=l&15, half=l>>4: elems 0..7 -> k = half*8 + s; elems 8..15 -> k = 16 + half*8 + (s-8).
__device__ __forceinline__ v16bf load_b_frag(const unsigned short* w, int ldk) {
    int l = (int)(threadIdx.x & 31);
    int n = l & 15, half = l >> 4;
    const unsigned short* base = w + n * ldk + half * 8;
    return load_frag16(base, base + 16);
}

// Generic tiled GEMM: C[16*mtiles][16*ntiles] = A[16*mtiles][32*ksteps] * W^T.
// Each wave owns whole (mtile,ntile) jobs -> wave-uniform loop, EXEC all ones at WMMA.
template <typename EMIT>
__device__ __forceinline__ void gemm_bf16(const unsigned short* A, int lda, int mtiles,
                                          const unsigned short* __restrict__ Wt, int ldk,
                                          int ntiles, int ksteps, EMIT emit) {
    int wave   = (int)(threadIdx.x >> 5);
    int nwaves = (int)(blockDim.x >> 5);
    int l      = (int)(threadIdx.x & 31);
    int n_l    = l & 15, half = l >> 4;
    int jobs   = mtiles * ntiles;
    for (int job = wave; job < jobs; job += nwaves) {
        int mt = job / ntiles, nt = job - mt * ntiles;
        v8f acc = {0.f, 0.f, 0.f, 0.f, 0.f, 0.f, 0.f, 0.f};
        const unsigned short* a0 = A  + mt * 16 * lda;
        const unsigned short* w0 = Wt + nt * 16 * ldk;
        for (int ks = 0; ks < ksteps; ++ks) {
            v16bf af = load_a_frag(a0 + ks * 32, lda);
            v16bf bfg = load_b_frag(w0 + ks * 32, ldk);
            acc = __builtin_amdgcn_wmma_f32_16x16x32_bf16(
                false, af, false, bfg, (short)0, acc, false, false);
        }
#pragma unroll
        for (int j = 0; j < 8; ++j)
            emit(mt * 16 + j + 8 * half, nt * 16 + n_l, acc[j]);
    }
}

// ---------------------------------------------------------------------------
// Weight conversion: f32 [K][N] row-major -> bf16 [Npad][Kpad] N-major, zero pad.
// splitK/gap lets us insert a zero gap (proj_w1: fused block moved to col 288).
// ---------------------------------------------------------------------------
__global__ void cvt_weight(const float* __restrict__ W, int Kd, int N,
                           unsigned short* __restrict__ Wt, int Kpad, int Npad,
                           int splitK, int gap) {
    int i = blockIdx.x * blockDim.x + threadIdx.x;
    if (i >= Kpad * Npad) return;
    int n = i / Kpad, k = i - n * Kpad;
    float v = 0.f;
    if (k < splitK) {
        if (k < Kd && n < N) v = W[(size_t)k * N + n];
    } else if (k >= splitK + gap) {
        int ks = k - gap;
        if (ks < Kd && n < N) v = W[(size_t)ks * N + n];
    }
    Wt[i] = f2bf(v);
}

// ---------------------------------------------------------------------------
// layer0: out = relu(concat(s,te) @ Wf) -> feat=[s,te,fused] -> relu(@W1) -> @W2
// 16 rows / block. feat LDS layout: [0,272) s|te, [272,288) zero, [288,460) fused.
// ---------------------------------------------------------------------------
struct L0Args {
    const int* ids; const float* tvals; int nrows;
    const float* spatial; const float* time_w; const float* time_b;
    const unsigned short *fus_t, *p1_t, *p2_t;
    const float *fus_b, *p1_b, *p2_b;
    float* out_f32;            // optional
    unsigned short* out_bf16;  // optional
};

__global__ __launch_bounds__(256) void layer0_kernel(L0Args a) {
    __shared__ __align__(32) unsigned short feat[16 * 480];
    __shared__ __align__(32) unsigned short hbuf[16 * 192];
    int tid  = (int)threadIdx.x;
    int row0 = (int)blockIdx.x * 16;

    for (int i = tid; i < 16 * 480; i += 256) feat[i] = 0;
    for (int i = tid; i < 16 * 192; i += 256) hbuf[i] = 0;
    __syncthreads();

    for (int i = tid; i < 16 * 272; i += 256) {
        int r = i / 272, c = i - r * 272;
        int row = row0 + r;
        float v;
        if (c < 172) v = a.spatial[(size_t)a.ids[row] * 172 + c];
        else { int j = c - 172; v = cosf(a.tvals[row] * a.time_w[j] + a.time_b[j]); }
        feat[r * 480 + c] = f2bf(v);
    }
    __syncthreads();

    // fused = relu([s,te] @ Wf + bf): reads cols [0,288), writes cols [288,460)
    gemm_bf16(feat, 480, 1, a.fus_t, 288, 11, 9, [&](int m, int n, float v) {
        if (n < 172) { v += a.fus_b[n]; feat[m * 480 + 288 + n] = f2bf(v > 0.f ? v : 0.f); }
    });
    __syncthreads();

    gemm_bf16(feat, 480, 1, a.p1_t, 480, 11, 15, [&](int m, int n, float v) {
        if (n < 172) { v += a.p1_b[n]; hbuf[m * 192 + n] = f2bf(v > 0.f ? v : 0.f); }
    });
    __syncthreads();

    gemm_bf16(hbuf, 192, 1, a.p2_t, 192, 11, 6, [&](int m, int n, float v) {
        if (n < 172) {
            v += a.p2_b[n];
            size_t o = (size_t)(row0 + m) * 172 + n;
            if (a.out_f32)  a.out_f32[o]  = v;
            if (a.out_bf16) a.out_bf16[o] = f2bf(v);
        }
    });
}

// ---------------------------------------------------------------------------
// conv: MHA (H=2, DH=86, K=20 nbrs) + residual + LayerNorm + MergeLayer MLP.
// 4 node rows / block => 80 kv rows = 5 exact WMMA M-tiles.
// ---------------------------------------------------------------------------
struct ConvArgs {
    int Brows;
    const float* node_f; const float* tvals;
    const int* nids; const int* eids; const float* ntimes;
    const float* nbr_f32; const unsigned short* nbr_bf16;   // one of them set
    const float* edge_feats; const float* time_w; const float* time_b;
    const unsigned short *wq_t, *wk_t, *wv_t, *wr_t, *m1_t, *m2_t;
    const float *br, *g, *beta, *b1, *b2;
    float* out;
};

__global__ __launch_bounds__(256) void conv_kernel(ConvArgs a) {
    __shared__ __align__(32) unsigned short kvb[80 * 448];  // kv input tile (bf16)
    __shared__ __align__(32) unsigned short kvo[80 * 176];  // K then V output (bf16)
    __shared__ __align__(32) unsigned short abuf[16 * 448]; // qin / attn-out / merge-in
    __shared__ __align__(32) unsigned short h1b[16 * 192];  // merge hidden (bf16)
    __shared__ float qin_f[4 * 272];
    __shared__ float qv[4 * 176];
    __shared__ float outv[4 * 272];
    __shared__ float sc[4 * 2 * 20];
    __shared__ float lnst[4 * 2];

    int tid  = (int)threadIdx.x;
    int row0 = (int)blockIdx.x * 4;

    for (int i = tid; i < 16 * 448; i += 256) abuf[i] = 0;
    for (int i = tid; i < 16 * 192; i += 256) h1b[i] = 0;
    for (int i = tid; i < 80 * 4; i += 256) { int r = i >> 2; kvb[r * 448 + 444 + (i & 3)] = 0; }
    __syncthreads();

    // ---- build q_in (node_f | tenc(0)=cos(b)) and kv (nbr_f | edge | tenc(dt)) ----
    for (int i = tid; i < 4 * 272; i += 256) {
        int r = i / 272, c = i - r * 272;
        float v = (c < 172) ? a.node_f[(size_t)(row0 + r) * 172 + c]
                            : cosf(a.time_b[c - 172]);
        qin_f[i] = v;
        abuf[r * 448 + c] = f2bf(v);
    }
    for (int i = tid; i < 80 * 444; i += 256) {
        int kr = i / 444, c = i - kr * 444;
        int r = kr / 20, k = kr - r * 20;
        int grow = row0 + r, gk = grow * 20 + k;
        float v;
        if (c < 172)
            v = a.nbr_bf16 ? bf2f(a.nbr_bf16[(size_t)gk * 172 + c])
                           : a.nbr_f32[(size_t)gk * 172 + c];
        else if (c < 344)
            v = a.edge_feats[(size_t)a.eids[gk] * 172 + (c - 172)];
        else {
            int j = c - 344;
            float dt = a.tvals[grow] - a.ntimes[gk];
            v = cosf(dt * a.time_w[j] + a.time_b[j]);
        }
        kvb[kr * 448 + c] = f2bf(v);
    }
    __syncthreads();

    // ---- Q projection + K projection ----
    gemm_bf16(abuf, 448, 1, a.wq_t, 288, 11, 9, [&](int m, int n, float v) {
        if (m < 4 && n < 172) qv[m * 176 + n] = v;
    });
    gemm_bf16(kvb, 448, 5, a.wk_t, 448, 11, 14, [&](int m, int n, float v) {
        kvo[m * 176 + n] = f2bf(v);
    });
    __syncthreads();

    // ---- scores (+ recycle abuf) ----
    if (tid < 80) {
        int r = tid / 20, k = tid - 20 * (tid / 20);
        int gk = (row0 + r) * 20 + k;
        float s0 = 0.f, s1 = 0.f;
        for (int d = 0; d < 86; ++d) {
            s0 += qv[r * 176 + d]      * bf2f(kvo[(r * 20 + k) * 176 + d]);
            s1 += qv[r * 176 + 86 + d] * bf2f(kvo[(r * 20 + k) * 176 + 86 + d]);
        }
        float scale = 0.10783277320343841f;  // 86^-0.5
        bool msk = (a.nids[gk] == 0);
        sc[(r * 2 + 0) * 20 + k] = msk ? -1e10f : s0 * scale;
        sc[(r * 2 + 1) * 20 + k] = msk ? -1e10f : s1 * scale;
    }
    for (int i = tid; i < 16 * 448; i += 256) abuf[i] = 0;
    __syncthreads();

    if (tid < 8) {  // softmax over 20 per (row,head)
        float* row = &sc[tid * 20];
        float mx = -3.4e38f;
        for (int k = 0; k < 20; ++k) mx = fmaxf(mx, row[k]);
        float s = 0.f;
        for (int k = 0; k < 20; ++k) { float e = expf(row[k] - mx); row[k] = e; s += e; }
        float inv = 1.f / s;
        for (int k = 0; k < 20; ++k) row[k] *= inv;
    }
    __syncthreads();

    // ---- V projection (overwrites kvo) ----
    gemm_bf16(kvb, 448, 5, a.wv_t, 448, 11, 14, [&](int m, int n, float v) {
        kvo[m * 176 + n] = f2bf(v);
    });
    __syncthreads();

    // ---- weighted sum -> abuf (bf16 tile for output projection) ----
    for (int i = tid; i < 4 * 172; i += 256) {
        int r = i / 172, c = i - 172 * r;
        int h = c / 86;
        float acc = 0.f;
        for (int k = 0; k < 20; ++k)
            acc += sc[(r * 2 + h) * 20 + k] * bf2f(kvo[(r * 20 + k) * 176 + c]);
        abuf[r * 448 + c] = f2bf(acc);
    }
    __syncthreads();

    // ---- out @ Wr + br + residual(q_in) ----
    gemm_bf16(abuf, 448, 1, a.wr_t, 192, 17, 6, [&](int m, int n, float v) {
        if (m < 4 && n < 272) outv[m * 272 + n] = v + a.br[n] + qin_f[m * 272 + n];
    });
    __syncthreads();

    // ---- LayerNorm ----
    if (tid < 4) {
        float mu = 0.f;
        for (int c = 0; c < 272; ++c) mu += outv[tid * 272 + c];
        mu *= (1.f / 272.f);
        float var = 0.f;
        for (int c = 0; c < 272; ++c) { float d = outv[tid * 272 + c] - mu; var += d * d; }
        var *= (1.f / 272.f);
        lnst[tid * 2] = mu; lnst[tid * 2 + 1] = rsqrtf(var + 1e-5f);
    }
    for (int i = tid; i < 16 * 448; i += 256) abuf[i] = 0;
    __syncthreads();

    for (int i = tid; i < 4 * 272; i += 256) {
        int r = i / 272, c = i - 272 * r;
        float v = (outv[i] - lnst[r * 2]) * lnst[r * 2 + 1] * a.g[c] + a.beta[c];
        outv[i] = v;
        abuf[r * 448 + c] = f2bf(v);
    }
    for (int i = tid; i < 4 * 172; i += 256) {
        int r = i / 172, c = i - 172 * r;
        abuf[r * 448 + 272 + c] = f2bf(a.node_f[(size_t)(row0 + r) * 172 + c]);
    }
    __syncthreads();

    // ---- MergeLayer: relu([out,node_f] @ W1 + b1) @ W2 + b2 ----
    gemm_bf16(abuf, 448, 1, a.m1_t, 448, 11, 14, [&](int m, int n, float v) {
        if (n < 172) { v += a.b1[n]; h1b[m * 192 + n] = f2bf(v > 0.f ? v : 0.f); }
    });
    __syncthreads();

    gemm_bf16(h1b, 192, 1, a.m2_t, 192, 11, 6, [&](int m, int n, float v) {
        if (m < 4 && n < 172)
            a.out[(size_t)(row0 + m) * 172 + n] = v + a.b2[n];
    });
}

// ---------------------------------------------------------------------------
// Host orchestration
// ---------------------------------------------------------------------------
extern "C" void kernel_launch(void* const* d_in, const int* in_sizes, int n_in,
                              void* d_out, int out_size, void* d_ws, size_t ws_size,
                              hipStream_t stream) {
    const int*   node_ids   = (const int*)d_in[0];
    const float* times      = (const float*)d_in[1];
    const int*   nbr1_ids   = (const int*)d_in[2];
    const int*   nbr1_eids  = (const int*)d_in[3];
    const float* nbr1_times = (const float*)d_in[4];
    const int*   nbr2_ids   = (const int*)d_in[5];
    const int*   nbr2_eids  = (const int*)d_in[6];
    const float* nbr2_times = (const float*)d_in[7];
    const float* spatial    = (const float*)d_in[8];
    const float* edgef      = (const float*)d_in[9];
    const float* time_w     = (const float*)d_in[10];
    const float* time_b     = (const float*)d_in[11];
    const float* fusion_w   = (const float*)d_in[12];
    const float* fusion_b   = (const float*)d_in[13];
    const float* proj_w1    = (const float*)d_in[14];
    const float* proj_b1    = (const float*)d_in[15];
    const float* proj_w2    = (const float*)d_in[16];
    const float* proj_b2    = (const float*)d_in[17];
    const float* attn_wq    = (const float*)d_in[18];
    const float* attn_wk    = (const float*)d_in[19];
    const float* attn_wv    = (const float*)d_in[20];
    const float* attn_wr    = (const float*)d_in[21];
    const float* attn_br    = (const float*)d_in[22];
    const float* attn_g     = (const float*)d_in[23];
    const float* attn_beta  = (const float*)d_in[24];
    const float* merge_w1   = (const float*)d_in[25];
    const float* merge_b1   = (const float*)d_in[26];
    const float* merge_w2   = (const float*)d_in[27];
    const float* merge_b2   = (const float*)d_in[28];
    (void)in_sizes; (void)n_in; (void)out_size; (void)ws_size;

    char* ws = (char*)d_ws;
    size_t off = 0;
    auto aU16 = [&](size_t n) -> unsigned short* {
        unsigned short* p = (unsigned short*)(ws + off);
        off += ((n * 2 + 255) / 256) * 256; return p;
    };
    auto aF32 = [&](size_t n) -> float* {
        float* p = (float*)(ws + off);
        off += ((n * 4 + 255) / 256) * 256; return p;
    };

    unsigned short* fus_t   = aU16(176 * 288);
    unsigned short* p1_t    = aU16(176 * 480);
    unsigned short* p2_t    = aU16(176 * 192);
    unsigned short* wq_t[2] = { aU16(176 * 288), aU16(176 * 288) };
    unsigned short* wk_t[2] = { aU16(176 * 448), aU16(176 * 448) };
    unsigned short* wv_t[2] = { aU16(176 * 448), aU16(176 * 448) };
    unsigned short* wr_t[2] = { aU16(272 * 192), aU16(272 * 192) };
    unsigned short* m1_t[2] = { aU16(176 * 448), aU16(176 * 448) };
    unsigned short* m2_t[2] = { aU16(176 * 192), aU16(176 * 192) };

    float* f_top0 = aF32((size_t)512 * 172);
    float* f_top1 = aF32((size_t)512 * 172);
    float* f_n1_0 = aF32((size_t)10240 * 172);
    float* f_n1_1 = aF32((size_t)10240 * 172);
    unsigned short* f_n2_0 = aU16((size_t)204800 * 172);

    auto cvt = [&](const float* src, int Kd, int N, unsigned short* dst,
                   int Kpad, int Npad, int splitK, int gap) {
        int tot = Kpad * Npad;
        cvt_weight<<<(tot + 255) / 256, 256, 0, stream>>>(src, Kd, N, dst, Kpad, Npad, splitK, gap);
    };
    const int NOSPLIT = 1 << 30;
    cvt(fusion_w, 272, 172, fus_t, 288, 176, NOSPLIT, 0);
    cvt(proj_w1,  444, 172, p1_t,  480, 176, 272, 16);   // fused block -> col 288
    cvt(proj_w2,  172, 172, p2_t,  192, 176, NOSPLIT, 0);
    for (int li = 0; li < 2; ++li) {
        cvt(attn_wq + (size_t)li * 272 * 172, 272, 172, wq_t[li], 288, 176, NOSPLIT, 0);
        cvt(attn_wk + (size_t)li * 444 * 172, 444, 172, wk_t[li], 448, 176, NOSPLIT, 0);
        cvt(attn_wv + (size_t)li * 444 * 172, 444, 172, wv_t[li], 448, 176, NOSPLIT, 0);
        cvt(attn_wr + (size_t)li * 172 * 272, 172, 272, wr_t[li], 192, 272, NOSPLIT, 0);
        cvt(merge_w1 + (size_t)li * 444 * 172, 444, 172, m1_t[li], 448, 176, NOSPLIT, 0);
        cvt(merge_w2 + (size_t)li * 172 * 172, 172, 172, m2_t[li], 192, 176, NOSPLIT, 0);
    }

    auto runL0 = [&](const int* ids, const float* tv, int nrows,
                     float* of32, unsigned short* obf) {
        L0Args a;
        a.ids = ids; a.tvals = tv; a.nrows = nrows;
        a.spatial = spatial; a.time_w = time_w; a.time_b = time_b;
        a.fus_t = fus_t; a.p1_t = p1_t; a.p2_t = p2_t;
        a.fus_b = fusion_b; a.p1_b = proj_b1; a.p2_b = proj_b2;
        a.out_f32 = of32; a.out_bf16 = obf;
        layer0_kernel<<<nrows / 16, 256, 0, stream>>>(a);
    };
    auto runConv = [&](int li, int B, const float* node_f, const float* tv,
                       const int* nids, const int* eids, const float* ntimes,
                       const float* nbrf32, const unsigned short* nbrbf, float* out) {
        ConvArgs a;
        a.Brows = B; a.node_f = node_f; a.tvals = tv;
        a.nids = nids; a.eids = eids; a.ntimes = ntimes;
        a.nbr_f32 = nbrf32; a.nbr_bf16 = nbrbf;
        a.edge_feats = edgef; a.time_w = time_w; a.time_b = time_b;
        a.wq_t = wq_t[li]; a.wk_t = wk_t[li]; a.wv_t = wv_t[li];
        a.wr_t = wr_t[li]; a.m1_t = m1_t[li]; a.m2_t = m2_t[li];
        a.br = attn_br + (size_t)li * 272; a.g = attn_g + (size_t)li * 272;
        a.beta = attn_beta + (size_t)li * 272;
        a.b1 = merge_b1 + (size_t)li * 172; a.b2 = merge_b2 + (size_t)li * 172;
        a.out = out;
        conv_kernel<<<B / 4, 256, 0, stream>>>(a);
    };

    // layer-0 embeddings
    runL0(node_ids, times, 512, f_top0, nullptr);
    runL0(nbr1_ids, nbr1_times, 10240, f_n1_0, nullptr);
    runL0(nbr2_ids, nbr2_times, 204800, nullptr, f_n2_0);

    // f_top1 = conv(0, top nodes over nbr1)
    runConv(0, 512, f_top0, times, nbr1_ids, nbr1_eids, nbr1_times,
            f_n1_0, nullptr, f_top1);
    // f_n1_1 = conv(0, nbr1 nodes over nbr2)
    runConv(0, 10240, f_n1_0, nbr1_times, nbr2_ids, nbr2_eids, nbr2_times,
            nullptr, f_n2_0, f_n1_1);
    // final = conv(1, top nodes over nbr1 with f_n1_1) -> d_out (src||dst contiguous)
    runConv(1, 512, f_top1, times, nbr1_ids, nbr1_eids, nbr1_times,
            f_n1_1, nullptr, (float*)d_out);
}
